// entropy_63737314673124
// MI455X (gfx1250) — matrix operand order repaired and checked by reading, not compiled, not run on hardware
//
#include <hip/hip_runtime.h>
#include <math.h>

// ---------------------------------------------------------------------------
// Problem constants (fixed by the reference: ex/ey are [8192, 512] fp32)
// ---------------------------------------------------------------------------
#define NROWS 8192
#define DIM   512
#define BT    128     // C tile edge per workgroup (128x128)
#define BK    32      // K chunk staged in LDS (one bf16 WMMA K-step)
#define PITCH 40      // LDS row pitch in bf16 elements (32 + 8 pad = 80B rows)

typedef __attribute__((ext_vector_type(16))) __bf16 v16bf;
typedef __attribute__((ext_vector_type(4)))  __bf16 v4bf;
typedef __attribute__((ext_vector_type(8)))  float  v8f;

union FragU { uint4 q[2]; v16bf v; };

__device__ __forceinline__ v16bf load_frag(const __bf16* s0, const __bf16* s1) {
    FragU f;
    f.q[0] = *(const uint4*)s0;   // VGPRs 0-3
    f.q[1] = *(const uint4*)s1;   // VGPRs 4-7
    return f.v;
}

// Async DMA: 16B per lane, global -> LDS, no VGPR round trip (ASYNCcnt).
// Low 32 bits of a generic LDS pointer == LDS byte offset (ISA aperture rule).
__device__ __forceinline__ void async_copy16(const void* g, void* l) {
    asm volatile("global_load_async_to_lds_b128 %0, %1, off"
                 :
                 : "v"((unsigned)(unsigned long long)l),
                   "v"((unsigned long long)g)
                 : "memory");
}
__device__ __forceinline__ void wait_async0() {
    asm volatile("s_wait_asynccnt 0" ::: "memory");
}

// Monotonic float -> uint encoding: order-preserving, so unsigned atomicMax
// implements exact float max. 0u is below every encoded finite float.
__device__ __forceinline__ unsigned encf(float f) {
    unsigned u = __float_as_uint(f);
    return (u & 0x80000000u) ? ~u : (u | 0x80000000u);
}
__device__ __forceinline__ float decf(unsigned u) {
    return (u & 0x80000000u) ? __uint_as_float(u ^ 0x80000000u)
                             : __uint_as_float(~u);
}

// ---------------------------------------------------------------------------
// Kernel 1: row L2-normalize + hi/lo bf16 decomposition.
//   x = hi + lo with hi = bf16_rne(x), lo = bf16_rne(x - hi).
// ---------------------------------------------------------------------------
__global__ __launch_bounds__(128) void normalize_split(
        const float* __restrict__ in,
        __bf16* __restrict__ hi, __bf16* __restrict__ lo) {
    const int row = blockIdx.x;
    const float4* src = (const float4*)(in + (size_t)row * DIM);
    float4 v = src[threadIdx.x];                       // 128 thr * 4 = 512
    float ss = v.x * v.x + v.y * v.y + v.z * v.z + v.w * v.w;
    #pragma unroll
    for (int m = 1; m < 32; m <<= 1) ss += __shfl_xor(ss, m, 32);
    __shared__ float wsum[4];
    if ((threadIdx.x & 31) == 0) wsum[threadIdx.x >> 5] = ss;
    __syncthreads();
    const float tot = wsum[0] + wsum[1] + wsum[2] + wsum[3];
    const float scale = 1.0f / fmaxf(sqrtf(tot), 1e-8f);
    v.x *= scale; v.y *= scale; v.z *= scale; v.w *= scale;

    v4bf h, l;
    h[0] = (__bf16)v.x; l[0] = (__bf16)(v.x - (float)h[0]);
    h[1] = (__bf16)v.y; l[1] = (__bf16)(v.y - (float)h[1]);
    h[2] = (__bf16)v.z; l[2] = (__bf16)(v.z - (float)h[2]);
    h[3] = (__bf16)v.w; l[3] = (__bf16)(v.w - (float)h[3]);
    *(v4bf*)&hi[(size_t)row * DIM + threadIdx.x * 4] = h;
    *(v4bf*)&lo[(size_t)row * DIM + threadIdx.x * 4] = l;
}

// ---------------------------------------------------------------------------
// Kernel 2: clear encoded max arrays (rowMax | colMax contiguous, 16384 u32)
// ---------------------------------------------------------------------------
__global__ void init_max(unsigned* __restrict__ p) {
    p[blockIdx.x * blockDim.x + threadIdx.x] = 0u;
}

// ---------------------------------------------------------------------------
// Kernel 3: split-bf16 WMMA GEMM tile + fused row/col max reduction.
//   dot(x,y) ~ hi_x.hi_y + hi_x.lo_y + lo_x.hi_y  (3 bf16 WMMAs per K=32)
//   256 threads = 8 waves in 2x4 grid; wave tile 64(M) x 32(N) = 4x2 WMMAs.
//   Global->LDS via async DMA (ASYNCcnt), double-buffered.
// ---------------------------------------------------------------------------
__global__ __launch_bounds__(256) void gemm_max(
        const __bf16* __restrict__ Xhi, const __bf16* __restrict__ Xlo,
        const __bf16* __restrict__ Yhi, const __bf16* __restrict__ Ylo,
        unsigned* __restrict__ rowMaxG, unsigned* __restrict__ colMaxG) {
    __shared__ __bf16 ldsAhi[2][BT * PITCH];   // X rows (M major)
    __shared__ __bf16 ldsAlo[2][BT * PITCH];
    __shared__ __bf16 ldsBhi[2][BT * PITCH];   // Y rows (N major)
    __shared__ __bf16 ldsBlo[2][BT * PITCH];
    __shared__ unsigned rowS[BT], colS[BT];

    const int tid     = threadIdx.x;
    const int rowBase = blockIdx.y * BT;    // M (rows of X / rows of C)
    const int colBase = blockIdx.x * BT;    // N (rows of Y / cols of C)

    const int lane = tid & 31;
    const int wave = tid >> 5;
    const int wm   = (wave & 1) * 64;       // wave M offset inside tile
    const int wn   = (wave >> 1) * 32;      // wave N offset inside tile
    const int half = lane >> 4;             // 0: lanes 0-15, 1: lanes 16-31
    const int lrow = lane & 15;

    if (tid < BT) { rowS[tid] = 0u; colS[tid] = 0u; }

    // Loader mapping: per plane, 128 rows x 4 (16B units of 8 bf16) = 512
    // units; 256 threads handle 2 units each.
    const int u0r = (tid + 0)   >> 2, u0c = ((tid + 0)   & 3) * 8;
    const int u1r = (tid + 256) >> 2, u1c = ((tid + 256) & 3) * 8;

    // ---- preload K-chunk 0 into buffer 0 (async DMA) --------------------
    {
        async_copy16(&Xhi[(size_t)(rowBase + u0r) * DIM + u0c], &ldsAhi[0][u0r * PITCH + u0c]);
        async_copy16(&Xhi[(size_t)(rowBase + u1r) * DIM + u1c], &ldsAhi[0][u1r * PITCH + u1c]);
        async_copy16(&Xlo[(size_t)(rowBase + u0r) * DIM + u0c], &ldsAlo[0][u0r * PITCH + u0c]);
        async_copy16(&Xlo[(size_t)(rowBase + u1r) * DIM + u1c], &ldsAlo[0][u1r * PITCH + u1c]);
        async_copy16(&Yhi[(size_t)(colBase + u0r) * DIM + u0c], &ldsBhi[0][u0r * PITCH + u0c]);
        async_copy16(&Yhi[(size_t)(colBase + u1r) * DIM + u1c], &ldsBhi[0][u1r * PITCH + u1c]);
        async_copy16(&Ylo[(size_t)(colBase + u0r) * DIM + u0c], &ldsBlo[0][u0r * PITCH + u0c]);
        async_copy16(&Ylo[(size_t)(colBase + u1r) * DIM + u1c], &ldsBlo[0][u1r * PITCH + u1c]);
    }

    const v8f vzero = {0.f,0.f,0.f,0.f,0.f,0.f,0.f,0.f};
    v8f acc[4][2];
    #pragma unroll
    for (int i = 0; i < 4; ++i)
        #pragma unroll
        for (int j = 0; j < 2; ++j) acc[i][j] = vzero;

    // ---- K loop: double-buffered LDS; async prefetch fills idle buffer --
    for (int kc = 0; kc < DIM; kc += BK) {
        const int buf = (kc / BK) & 1;
        wait_async0();                      // own async fills landed in LDS
        __syncthreads();                    // ... and everyone else's too

        const int kn = kc + BK;
        if (kn < DIM) {
            const int nb = buf ^ 1;
            async_copy16(&Xhi[(size_t)(rowBase + u0r) * DIM + kn + u0c], &ldsAhi[nb][u0r * PITCH + u0c]);
            async_copy16(&Xhi[(size_t)(rowBase + u1r) * DIM + kn + u1c], &ldsAhi[nb][u1r * PITCH + u1c]);
            async_copy16(&Xlo[(size_t)(rowBase + u0r) * DIM + kn + u0c], &ldsAlo[nb][u0r * PITCH + u0c]);
            async_copy16(&Xlo[(size_t)(rowBase + u1r) * DIM + kn + u1c], &ldsAlo[nb][u1r * PITCH + u1c]);
            async_copy16(&Yhi[(size_t)(colBase + u0r) * DIM + kn + u0c], &ldsBhi[nb][u0r * PITCH + u0c]);
            async_copy16(&Yhi[(size_t)(colBase + u1r) * DIM + kn + u1c], &ldsBhi[nb][u1r * PITCH + u1c]);
            async_copy16(&Ylo[(size_t)(colBase + u0r) * DIM + kn + u0c], &ldsBlo[nb][u0r * PITCH + u0c]);
            async_copy16(&Ylo[(size_t)(colBase + u1r) * DIM + kn + u1c], &ldsBlo[nb][u1r * PITCH + u1c]);
        }

        const __bf16* __restrict__ Ah = ldsAhi[buf];
        const __bf16* __restrict__ Al = ldsAlo[buf];
        const __bf16* __restrict__ Bh = ldsBhi[buf];
        const __bf16* __restrict__ Bl = ldsBlo[buf];

        // B fragments: 32x16 bf16, lane holds N=lrow, K = 16*half..+15
        v16bf bhi[2], blo[2];
        #pragma unroll
        for (int j = 0; j < 2; ++j) {
            const int br = (wn + j * 16 + lrow) * PITCH + 16 * half;
            bhi[j] = load_frag(&Bh[br], &Bh[br + 8]);
            blo[j] = load_frag(&Bl[br], &Bl[br + 8]);
        }
        // A fragments: 16x32 bf16, lane holds M=lrow,
        // K = 16*half..+15 (VGPR0-7)
        #pragma unroll
        for (int i = 0; i < 4; ++i) {
            const int ar = (wm + i * 16 + lrow) * PITCH + 16 * half;
            v16bf ahi = load_frag(&Ah[ar], &Ah[ar + 8]);
            v16bf alo = load_frag(&Al[ar], &Al[ar + 8]);
            #pragma unroll
            for (int j = 0; j < 2; ++j) {
                acc[i][j] = __builtin_amdgcn_wmma_f32_16x16x32_bf16(
                    false, ahi, false, bhi[j], (short)0, acc[i][j], false, false);
                acc[i][j] = __builtin_amdgcn_wmma_f32_16x16x32_bf16(
                    false, ahi, false, blo[j], (short)0, acc[i][j], false, false);
                acc[i][j] = __builtin_amdgcn_wmma_f32_16x16x32_bf16(
                    false, alo, false, bhi[j], (short)0, acc[i][j], false, false);
            }
        }
    }

    // ---- max reductions -------------------------------------------------
    // C/D layout: VGPR v, lane<16: C[M=v][N=lrow]; lane>=16: C[M=v+8][N=lrow].
    // Row max (over N): combine j-tiles, butterfly across the 16-lane half.
    #pragma unroll
    for (int i = 0; i < 4; ++i) {
        float rm[8];
        #pragma unroll
        for (int v = 0; v < 8; ++v) rm[v] = fmaxf(acc[i][0][v], acc[i][1][v]);
        #pragma unroll
        for (int m = 1; m < 16; m <<= 1)
            #pragma unroll
            for (int v = 0; v < 8; ++v)
                rm[v] = fmaxf(rm[v], __shfl_xor(rm[v], m, 32));
        #pragma unroll
        for (int v = 0; v < 8; ++v)
            if (lrow == v)
                atomicMax(&rowS[wm + i * 16 + half * 8 + v], encf(rm[v]));
    }
    // Col max (over M): max over 8 VGPRs and 4 i-tiles, then swap halves.
    #pragma unroll
    for (int j = 0; j < 2; ++j) {
        float cm = acc[0][j][0];
        #pragma unroll
        for (int i = 0; i < 4; ++i)
            #pragma unroll
            for (int v = 0; v < 8; ++v) cm = fmaxf(cm, acc[i][j][v]);
        cm = fmaxf(cm, __shfl_xor(cm, 16, 32));
        if (half == 0) atomicMax(&colS[wn + j * 16 + lrow], encf(cm));
    }
    __syncthreads();
    if (tid < BT) {
        atomicMax(&rowMaxG[rowBase + tid], rowS[tid]);
        atomicMax(&colMaxG[colBase + tid], colS[tid]);
    }
}

// ---------------------------------------------------------------------------
// Kernel 4: entropy terms.  lp = -0.5*((x-1)/0.3)^2 - log(0.3) - 0.5*log(2pi)
//           out = -sum(exp(lp)*lp)  for row-maxes (out[0]) and col-maxes (out[1])
// ---------------------------------------------------------------------------
__global__ __launch_bounds__(256) void finalize(
        const unsigned* __restrict__ rowMaxG,
        const unsigned* __restrict__ colMaxG, float* __restrict__ out) {
    const float INV_SIGMA = 1.0f / 0.3f;
    const float C0 = 0.28503428f;   // -log(0.3) - 0.5*log(2*pi)
    float sr = 0.f, sc = 0.f;
    for (int i = threadIdx.x; i < NROWS; i += 256) {
        float x = decf(rowMaxG[i]);
        float t = (x - 1.0f) * INV_SIGMA;
        float lp = -0.5f * t * t + C0;
        sr -= expf(lp) * lp;
        x = decf(colMaxG[i]);
        t = (x - 1.0f) * INV_SIGMA;
        lp = -0.5f * t * t + C0;
        sc -= expf(lp) * lp;
    }
    #pragma unroll
    for (int m = 1; m < 32; m <<= 1) {
        sr += __shfl_xor(sr, m, 32);
        sc += __shfl_xor(sc, m, 32);
    }
    __shared__ float red[2][8];
    if ((threadIdx.x & 31) == 0) {
        red[0][threadIdx.x >> 5] = sr;
        red[1][threadIdx.x >> 5] = sc;
    }
    __syncthreads();
    if (threadIdx.x == 0) {
        float a = 0.f, b = 0.f;
        #pragma unroll
        for (int w = 0; w < 8; ++w) { a += red[0][w]; b += red[1][w]; }
        out[0] = a;   // entropy of lp_right (row maxes, axis=-1)
        out[1] = b;   // entropy of lp_left  (col maxes, axis=0)
    }
}

// ---------------------------------------------------------------------------
extern "C" void kernel_launch(void* const* d_in, const int* in_sizes, int n_in,
                              void* d_out, int out_size, void* d_ws, size_t ws_size,
                              hipStream_t stream) {
    const float* ex = (const float*)d_in[0];
    const float* ey = (const float*)d_in[1];
    float* out = (float*)d_out;

    // Workspace: Xhi|Xlo|Yhi|Ylo (8MB each, bf16) | rowMax | colMax (u32)
    __bf16* Xhi = (__bf16*)d_ws;
    __bf16* Xlo = Xhi + (size_t)NROWS * DIM;
    __bf16* Yhi = Xlo + (size_t)NROWS * DIM;
    __bf16* Ylo = Yhi + (size_t)NROWS * DIM;
    unsigned* rowMax = (unsigned*)(Ylo + (size_t)NROWS * DIM);
    unsigned* colMax = rowMax + NROWS;

    normalize_split<<<NROWS, 128, 0, stream>>>(ex, Xhi, Xlo);
    normalize_split<<<NROWS, 128, 0, stream>>>(ey, Yhi, Ylo);
    init_max<<<(2 * NROWS) / 256, 256, 0, stream>>>(rowMax);

    dim3 grid(NROWS / BT, NROWS / BT);   // 64 x 64 workgroups
    gemm_max<<<grid, 256, 0, stream>>>(Xhi, Xlo, Yhi, Ylo, rowMax, colMax);

    finalize<<<1, 256, 0, stream>>>(rowMax, colMax, out);
}
